// Encoder_19473381720339
// MI455X (gfx1250) — matrix operand order repaired and checked by reading, not compiled, not run on hardware
//
#include <hip/hip_runtime.h>
#include <hip/hip_bf16.h>

typedef __attribute__((ext_vector_type(16))) __bf16 v16bf;
typedef __attribute__((ext_vector_type(8)))  __bf16 v8bf;
typedef __attribute__((ext_vector_type(8)))  float  v8f;
typedef __attribute__((ext_vector_type(4)))  float  v4f;

#define N_NODES 100000
#define N_EDGES 1600000
#define DIM     128
#define ND      (N_NODES * DIM)        // 12,800,000
#define ROW_TILES (N_NODES / 16)       // 6250 (exact)

// ---------------------------------------------------------------- utility ---
__global__ void k_fill_f32(float* __restrict__ p, float v, int n) {
    int i = blockIdx.x * blockDim.x + threadIdx.x;
    if (i < n) p[i] = v;
}

__global__ void k_cvt_bf16(const float* __restrict__ in, __bf16* __restrict__ out, int n) {
    int i = blockIdx.x * blockDim.x + threadIdx.x;
    if (i < n) out[i] = (__bf16)in[i];
}

// deg[src[e]] += 1  (deg pre-filled with 1.0 for the self loop)
__global__ void k_degree(const int* __restrict__ src, float* __restrict__ deg) {
    int e = blockIdx.x * blockDim.x + threadIdx.x;
    if (e < N_EDGES) atomicAdd(&deg[src[e]], 1.0f);
}

__global__ void k_dinv(float* __restrict__ deg, int n) {
    int i = blockIdx.x * blockDim.x + threadIdx.x;
    if (i < n) deg[i] = rsqrtf(deg[i]);     // in place: deg -> dinv
}

// h0 = leaky_relu(x @ fc_w.T + fc_b); also emit bf16 shadow for the GEMM
__global__ void k_fc(const float* __restrict__ x, const float* __restrict__ w,
                     const float* __restrict__ b, float* __restrict__ h,
                     __bf16* __restrict__ hbf) {
    int idx = blockIdx.x * blockDim.x + threadIdx.x;
    if (idx >= ND) return;
    int i = idx / DIM, d = idx % DIM;
    float s = b[d];
    s = fmaf(x[i * 3 + 0], w[d * 3 + 0], s);
    s = fmaf(x[i * 3 + 1], w[d * 3 + 1], s);
    s = fmaf(x[i * 3 + 2], w[d * 3 + 2], s);
    s = (s > 0.0f) ? s : 0.01f * s;
    h[idx]   = s;
    hbf[idx] = (__bf16)s;
}

// ------------------------------------------------------------- WMMA GEMM ---
// C[M, Dout] = A_bf16[M,128] x W_bf16[Dout,128]^T + bias[Dout]
// 4 waves / block; each wave -> 16 rows x 64 cols (4 accumulators).
__global__ __launch_bounds__(128) void k_gemm_bf16(
        const __bf16* __restrict__ A, const __bf16* __restrict__ W,
        const float* __restrict__ bias, float* __restrict__ C,
        int M, int Dout) {
    const int lane = threadIdx.x & 31;
    const int wave = threadIdx.x >> 5;
    const int row_tile = blockIdx.x * 4 + wave;
    const int row0 = row_tile * 16;
    if (row0 >= M) return;                      // wave-uniform: EXEC stays full
    const int col0 = blockIdx.y * 64;
    const int half = lane >> 4;                 // 0 or 1
    const int l16  = lane & 15;

    const __bf16* ap = A + (size_t)(row0 + l16) * DIM + half * 16;
    v8f acc[4] = {v8f{}, v8f{}, v8f{}, v8f{}};

#pragma unroll
    for (int kc = 0; kc < 4; ++kc) {            // K = 128 in chunks of 32
        // A fragment (16x32 bf16): lane holds 16 contiguous K = {16*half..16*half+15}
        v8bf alo = *(const v8bf*)(ap + kc * 32);
        v8bf ahi = *(const v8bf*)(ap + kc * 32 + 8);
        v16bf a;
#pragma unroll
        for (int i = 0; i < 8; ++i) { a[i] = alo[i]; a[8 + i] = ahi[i]; }

#pragma unroll
        for (int t = 0; t < 4; ++t) {
            // B fragment (32x16 bf16): lane = col, 16 contiguous K values
            const __bf16* bp = W + (size_t)(col0 + t * 16 + l16) * DIM
                                 + kc * 32 + half * 16;
            v16bf b = *(const v16bf*)bp;
            acc[t] = __builtin_amdgcn_wmma_f32_16x16x32_bf16(
                         false, a, false, b, (short)0, acc[t], false, false);
        }
    }

#pragma unroll
    for (int t = 0; t < 4; ++t) {
        const int col = col0 + t * 16 + l16;
        const float bv = bias[col];
#pragma unroll
        for (int r = 0; r < 8; ++r) {
            const int row = row0 + half * 8 + r;
            C[(size_t)row * Dout + col] = acc[t][r] + bv;
        }
    }
}

// ----------------------------------------------------- edge scatter (SpMM) ---
// one wave per edge (incl. N self loops); lane handles 4 dims (float4)
__global__ void k_scatter(const float* __restrict__ t, const int* __restrict__ src,
                          const int* __restrict__ dst, const float* __restrict__ dinv,
                          float* __restrict__ agg) {
    int tid  = blockIdx.x * blockDim.x + threadIdx.x;
    int e    = tid >> 5;
    int lane = tid & 31;
    if (e >= N_EDGES + N_NODES) return;
    int s, d;
    if (e < N_EDGES) { s = src[e]; d = dst[e]; }
    else             { s = d = e - N_EDGES; }
    const float w = dinv[s] * dinv[d];
    v4f v = *((const v4f*)(t + (size_t)s * DIM) + lane);
    float* op = agg + (size_t)d * DIM + lane * 4;
    atomicAdd(op + 0, w * v[0]);
    atomicAdd(op + 1, w * v[1]);
    atomicAdd(op + 2, w * v[2]);
    atomicAdd(op + 3, w * v[3]);
}

// ------------------------------------------------------------ GRU + max ---
__global__ void k_gru(const float* __restrict__ gi, const float* __restrict__ gh,
                      const float* __restrict__ h, float* __restrict__ hnew,
                      __bf16* __restrict__ hnew_bf, float* __restrict__ out,
                      int first) {
    int idx = blockIdx.x * blockDim.x + threadIdx.x;
    if (idx >= ND) return;
    int i = idx / DIM, d = idx % DIM;
    size_t base = (size_t)i * 384;
    float r = 1.0f / (1.0f + __expf(-(gi[base +       d] + gh[base +       d])));
    float z = 1.0f / (1.0f + __expf(-(gi[base + 128 + d] + gh[base + 128 + d])));
    float n = tanhf(gi[base + 256 + d] + r * gh[base + 256 + d]);
    float hp = h[idx];
    float hn = (1.0f - z) * n + z * hp;
    hnew[idx]    = hn;
    hnew_bf[idx] = (__bf16)hn;
    out[idx] = first ? hn : fmaxf(out[idx], hn);
}

// ------------------------------------------------------------------ host ---
extern "C" void kernel_launch(void* const* d_in, const int* in_sizes, int n_in,
                              void* d_out, int out_size, void* d_ws, size_t ws_size,
                              hipStream_t stream) {
    const float* x    = (const float*)d_in[0];
    const int*   ei   = (const int*)d_in[1];
    const int*   src  = ei;
    const int*   dst  = ei + N_EDGES;
    const float* fc_w = (const float*)d_in[2];
    const float* fc_b = (const float*)d_in[3];

    const float *gcn_w[3], *gcn_b[3], *wih[3], *whh[3], *bih[3], *bhh[3];
    for (int l = 0; l < 3; ++l) {
        int base = 4 + 6 * l;
        gcn_w[l] = (const float*)d_in[base + 0];
        gcn_b[l] = (const float*)d_in[base + 1];
        wih[l]   = (const float*)d_in[base + 2];
        whh[l]   = (const float*)d_in[base + 3];
        bih[l]   = (const float*)d_in[base + 4];
        bhh[l]   = (const float*)d_in[base + 5];
    }

    // ---- workspace carve (256B aligned) ----
    char* p = (char*)d_ws;
    auto carve = [&](size_t bytes) -> void* {
        void* r = (void*)p;
        p += (bytes + 255) & ~(size_t)255;
        return r;
    };
    float*  dinv    = (float*) carve((size_t)N_NODES * 4);
    float*  hA      = (float*) carve((size_t)ND * 4);
    __bf16* hA_bf   = (__bf16*)carve((size_t)ND * 2);
    float*  hB      = (float*) carve((size_t)ND * 4);
    __bf16* hB_bf   = (__bf16*)carve((size_t)ND * 2);
    float*  tlin    = (float*) carve((size_t)ND * 4);
    float*  agg     = (float*) carve((size_t)ND * 4);
    __bf16* agg_bf  = (__bf16*)carve((size_t)ND * 2);
    float*  gi      = (float*) carve((size_t)N_NODES * 384 * 4);
    float*  gh      = (float*) carve((size_t)N_NODES * 384 * 4);
    __bf16* gcnw_bf[3]; __bf16* wih_bf[3]; __bf16* whh_bf[3];
    for (int l = 0; l < 3; ++l) {
        gcnw_bf[l] = (__bf16*)carve((size_t)DIM * DIM * 2);
        wih_bf[l]  = (__bf16*)carve((size_t)3 * DIM * DIM * 2);
        whh_bf[l]  = (__bf16*)carve((size_t)3 * DIM * DIM * 2);
    }

    float* out = (float*)d_out;

    // ---- degree / norm ----
    k_fill_f32<<<(N_NODES + 255) / 256, 256, 0, stream>>>(dinv, 1.0f, N_NODES);
    k_degree<<<(N_EDGES + 255) / 256, 256, 0, stream>>>(src, dinv);
    k_dinv<<<(N_NODES + 255) / 256, 256, 0, stream>>>(dinv, N_NODES);

    // ---- weight conversions to bf16 ----
    for (int l = 0; l < 3; ++l) {
        k_cvt_bf16<<<(DIM * DIM + 255) / 256, 256, 0, stream>>>(gcn_w[l], gcnw_bf[l], DIM * DIM);
        k_cvt_bf16<<<(3 * DIM * DIM + 255) / 256, 256, 0, stream>>>(wih[l], wih_bf[l], 3 * DIM * DIM);
        k_cvt_bf16<<<(3 * DIM * DIM + 255) / 256, 256, 0, stream>>>(whh[l], whh_bf[l], 3 * DIM * DIM);
    }

    // ---- h0 = leaky_relu(x @ fc_w.T + fc_b) ----
    k_fc<<<(ND + 255) / 256, 256, 0, stream>>>(x, fc_w, fc_b, hA, hA_bf);

    float*  hcur    = hA;   __bf16* hcur_bf = hA_bf;
    float*  hnxt    = hB;   __bf16* hnxt_bf = hB_bf;

    const dim3 gemm_block(128);
    const int  gx = (ROW_TILES + 3) / 4;            // 1563
    const int  scat_threads = (N_EDGES + N_NODES) * 32;

    for (int l = 0; l < 3; ++l) {
        // t = h @ gcn_w.T + gcn_b
        k_gemm_bf16<<<dim3(gx, DIM / 64), gemm_block, 0, stream>>>(
            hcur_bf, gcnw_bf[l], gcn_b[l], tlin, N_NODES, DIM);

        // agg = segment_sum(norm * t[src], dst)
        hipMemsetAsync(agg, 0, (size_t)ND * 4, stream);
        k_scatter<<<(scat_threads + 255) / 256, 256, 0, stream>>>(
            tlin, src, dst, dinv, agg);
        k_cvt_bf16<<<(ND + 255) / 256, 256, 0, stream>>>(agg, agg_bf, ND);

        // gi = agg @ wih.T + bih ; gh = h @ whh.T + bhh
        k_gemm_bf16<<<dim3(gx, 384 / 64), gemm_block, 0, stream>>>(
            agg_bf, wih_bf[l], bih[l], gi, N_NODES, 384);
        k_gemm_bf16<<<dim3(gx, 384 / 64), gemm_block, 0, stream>>>(
            hcur_bf, whh_bf[l], bhh[l], gh, N_NODES, 384);

        // GRU gates + running max into d_out
        k_gru<<<(ND + 255) / 256, 256, 0, stream>>>(
            gi, gh, hcur, hnxt, hnxt_bf, out, l == 0 ? 1 : 0);

        // swap ping-pong
        float*  tf = hcur;    hcur    = hnxt;    hnxt    = tf;
        __bf16* tb = hcur_bf; hcur_bf = hnxt_bf; hnxt_bf = tb;
    }
}
